// Maj3_25933012533391
// MI455X (gfx1250) — 8-bit matrix operand order repaired and checked by reading, compiled, not run on hardware
//
#include <hip/hip_runtime.h>
#include <stdint.h>

typedef __attribute__((ext_vector_type(8))) int v8i;

#define B_ 8
#define W_ 32
#define H_ 32
#define CIN 64
#define COUT 128
#define J576 576
#define K192 192
#define KTOT 768
#define NROWS 8192

__global__ void maj3_pack_x(const float* __restrict__ x, int8_t* __restrict__ A) {
    int idx = blockIdx.x * blockDim.x + threadIdx.x;
    if (idx >= NROWS * J576) return;
    int n = idx / J576, j = idx - n * J576;
    int c = j / 9, r = j - c * 9;
    int kw = r / 3, kh = r - kw * 3;
    int b = n >> 10, wh = n & 1023, w = wh >> 5, h = wh & 31;
    int wi = w + kw - 1, hi = h + kh - 1;
    int8_t s;
    if (wi < 0 || wi >= W_ || hi < 0 || hi >= H_) {
        s = -1;
    } else {
        float v = x[((((b * W_) + wi) * H_ + hi) << 6) + c];
        s = (v > 0.0f) ? (int8_t)1 : ((v < 0.0f) ? (int8_t)-1 : (int8_t)0);
    }
    A[(size_t)n * KTOT + j] = s;
}

__global__ void maj3_pack_x3(int8_t* __restrict__ A) {
    int idx = blockIdx.x * blockDim.x + threadIdx.x;
    if (idx >= NROWS * K192) return;
    int n = idx / K192, k = idx - n * K192;
    int8_t* row = A + (size_t)n * KTOT;
    row[J576 + k] = (int8_t)((int)row[k] * (int)row[k + K192] * (int)row[k + 2 * K192]);
}

__global__ void maj3_pack_w(const float* __restrict__ w, int8_t* __restrict__ WS) {
    int a = blockIdx.x * blockDim.x + threadIdx.x;
    if (a >= COUT) return;
    int8_t* row = WS + (size_t)a * KTOT;
    for (int j = 0; j < J576; ++j) {
        float v = w[a * J576 + j];
        row[j] = (v > 0.0f) ? (int8_t)1 : ((v < 0.0f) ? (int8_t)-1 : (int8_t)0);
    }
    for (int k = 0; k < K192; ++k)
        row[J576 + k] = (int8_t)(-((int)row[k] * (int)row[k + K192] * (int)row[k + 2 * K192]));
}

__global__ __launch_bounds__(128) void maj3_gemm(
    const int8_t* __restrict__ A, const int8_t* __restrict__ WS,
    const float* __restrict__ bias, float* __restrict__ out) {

    __shared__ int4 lds_w4[32 * KTOT / 16];
    const int* lds_w = (const int*)lds_w4;

    int tid  = threadIdx.x;
    int wave = tid >> 5, lane = tid & 31;
    int half = lane >> 4, mlane = lane & 15;
    int m0 = (blockIdx.x * 4 + wave) * 16;

    const int* arow = (const int*)(A + (size_t)(m0 + mlane) * KTOT);

    int brow0 = mlane * (KTOT / 4) + (half << 2);
    int brow1 = brow0 + 16 * (KTOT / 4);

    for (int ntp = 0; ntp < COUT / 32; ++ntp) {
        __syncthreads();
        {
            const int4* src = (const int4*)(WS + (size_t)ntp * 32 * KTOT);
            for (int i = tid; i < 32 * KTOT / 16; i += 128) lds_w4[i] = src[i];
        }
        __syncthreads();

        v8i acc0 = {}, acc1 = {};
        #pragma unroll
        for (int kk = 0; kk < KTOT / 64; ++kk) {
            int kb = kk * 64;
            v8i af, bf0, bf1;
            #pragma unroll
            for (int d = 0; d < 8; ++d) {
                int ka = kb + 32 * (d >> 2) + ((d & 3) << 2) + (half << 4);
                af[d] = arow[ka >> 2];
                int kbo = (kb >> 2) + ((d >> 2) << 3) + (d & 3);
                bf0[d] = lds_w[brow0 + kbo];
                bf1[d] = lds_w[brow1 + kbo];
            }
            acc0 = __builtin_amdgcn_wmma_i32_16x16x64_iu8(true, af, true, bf0, acc0, false, false);
            acc1 = __builtin_amdgcn_wmma_i32_16x16x64_iu8(true, af, true, bf1, acc1, false, false);
        }

        int a0 = ntp * 32 + mlane;
        int a1 = a0 + 16;
        float bv0 = bias[a0];
        float bv1 = bias[a1];
        int bidx = m0 >> 10;
        int wh0  = (m0 & 1023) + (half << 3);
        float* ob0 = out + ((size_t)bidx << 17) + ((size_t)a0 << 10) + wh0;
        float* ob1 = out + ((size_t)bidx << 17) + ((size_t)a1 << 10) + wh0;
        #pragma unroll
        for (int r = 0; r < 8; ++r) {
            ob0[r] = 1.5f * (float)acc0[r] + bv0;
            ob1[r] = 1.5f * (float)acc1[r] + bv1;
        }
    }
}

extern "C" void kernel_launch(void* const* d_in, const int* in_sizes, int n_in,
                              void* d_out, int out_size, void* d_ws, size_t ws_size,
                              hipStream_t stream) {
    const float* x    = (const float*)d_in[0];
    const float* wgt  = (const float*)d_in[1];
    const float* bias = (const float*)d_in[2];
    float* out = (float*)d_out;

    int8_t* A  = (int8_t*)d_ws;
    int8_t* WS = A + (size_t)NROWS * KTOT;

    {
        int total = NROWS * J576;
        maj3_pack_x<<<(total + 255) / 256, 256, 0, stream>>>(x, A);
    }
    {
        int total = NROWS * K192;
        maj3_pack_x3<<<(total + 255) / 256, 256, 0, stream>>>(A);
    }
    maj3_pack_w<<<1, 128, 0, stream>>>(wgt, WS);

    maj3_gemm<<<NROWS / 64, 128, 0, stream>>>(A, WS, bias, out);
}
